// ModelPytorch_91190745628837
// MI455X (gfx1250) — matrix operand order repaired and checked by reading, not compiled, not run on hardware
//
#include <hip/hip_runtime.h>

#define HIDDEN 32
#define B_TOT  2048
#define T_MAIN 2048
#define T_EXTW 8
#define T_PRED 16
#define T_OUT  (T_MAIN + T_PRED)
#define TC     128           // t-chunk staged through LDS
#define XSTR   (TC + 4)      // x tile row stride (16 rows, broadcast reads; +pad for prefetch)
#define YSTR2  (TC + 5)      // partial-y stride: 133 -> all 32 rows conflict-free
#define PSTR   28            // ybuf row stride (24 used)

typedef __attribute__((ext_vector_type(16))) _Float16     v16h;
typedef __attribute__((ext_vector_type(8)))  float        v8f;
typedef __attribute__((ext_vector_type(8)))  unsigned int v8u;
typedef __attribute__((ext_vector_type(2)))  _Float16     v2h;

static __device__ __forceinline__ float fast_tanh(float x) {
#if __has_builtin(__builtin_amdgcn_tanhf)
  return __builtin_amdgcn_tanhf(x);          // gfx1250 native v_tanh_f32
#elif __has_builtin(__builtin_amdgcn_tanh_f32)
  return __builtin_amdgcn_tanh_f32(x);
#else
  float xc = fminf(fmaxf(x, -9.0f), 9.0f);
  float e  = __builtin_amdgcn_exp2f(xc * 2.885390081777927f); // exp(2x)
  return (e - 1.0f) * __builtin_amdgcn_rcpf(e + 1.0f);
#endif
}

static __device__ __forceinline__ unsigned swap16u(unsigned x) {
  // ds_swizzle group-of-32: xor_mask=0x10, and_mask=0x1f  -> lane ^= 16
  return (unsigned)__builtin_amdgcn_ds_swizzle((int)x, 0x401F);
}
static __device__ __forceinline__ float swap16f(float x) {
  return __builtin_bit_cast(float, swap16u(__builtin_bit_cast(unsigned, x)));
}
static __device__ __forceinline__ unsigned pkh2(float a, float b) {
  v2h p; p.x = (_Float16)a; p.y = (_Float16)b;
  return __builtin_bit_cast(unsigned, p);
}

__global__ __launch_bounds__(32) void rnn_wave_kernel(
    const float* __restrict__ x,      // [B, T] (IN_DIM=1 flattened)
    const float* __restrict__ W_ih,   // [32]
    const float* __restrict__ b_ih,   // [32]
    const float* __restrict__ W_hh,   // [32,32] row-major (out, in)
    const float* __restrict__ b_hh,   // [32]
    const float* __restrict__ W_dec,  // [32]
    const float* __restrict__ b_dec,  // [1]
    float* __restrict__ y)            // [B, T_OUT]
{
  __shared__ float xs [16 * XSTR];
  __shared__ float ys2[32 * YSTR2];   // per-lane decode partials (both halves)
  __shared__ float ybuf[16 * PSTR];   // last-8 window + 16 predictions per batch row

  const int lane = threadIdx.x & 31;
  const int n    = lane & 15;   // batch column within tile
  const int hi   = lane >> 4;   // lane half
  const int b0   = blockIdx.x * 16;

  // ---- W_hh into WMMA A layout (f16), with PERMUTED row tiles so D lands
  // exactly in B-operand lane halves (no cross-lane repack):
  //   A0 rows (M=0..15) -> hidden {0-7, 16-23}
  //   A1 rows (M=0..15) -> hidden {8-15, 24-31}
  // A k order used here (contiguous, 16x32): lane L: M=L%16; element e: K=e+(L>=16 ? 16:0)
  const int g0 = (n < 8) ? n : n + 8;        // A0 global row for this lane's M
  const int g1 = (n < 8) ? n + 8 : n + 16;   // A1 global row
  v8u a0u = {0,0,0,0,0,0,0,0}, a1u = {0,0,0,0,0,0,0,0};
#pragma unroll
  for (int r = 0; r < 8; ++r) {
    int e0 = 2 * r, e1 = 2 * r + 1;
    int K0 = e0 + (hi ? 16 : 0);
    int K1 = e1 + (hi ? 16 : 0);
    a0u[r] = pkh2(W_hh[g0 * 32 + K0], W_hh[g0 * 32 + K1]);
    a1u[r] = pkh2(W_hh[g1 * 32 + K0], W_hh[g1 * 32 + K1]);
  }
  const v16h A0 = __builtin_bit_cast(v16h, a0u);
  const v16h A1 = __builtin_bit_cast(v16h, a1u);

  // ---- per-lane scalars in C/D layout:
  //   D0 VGPR r -> hidden m0 = hi ? 16+r : r
  //   D1 VGPR r -> hidden m1 = hi ? 24+r : 8+r
  float wih0[8], wih1[8], bb0[8], bb1[8], wd0[8], wd1[8];
#pragma unroll
  for (int r = 0; r < 8; ++r) {
    int m0 = hi ? 16 + r : r;
    int m1 = hi ? 24 + r : 8 + r;
    wih0[r] = W_ih[m0];  wih1[r] = W_ih[m1];
    bb0[r]  = b_ih[m0] + b_hh[m0];
    bb1[r]  = b_ih[m1] + b_hh[m1];
    wd0[r]  = W_dec[m0]; wd1[r] = W_dec[m1];
  }
  const float bdec = b_dec[0];

  // H state: f16 pairs in B-operand layout (lanes 0-15: K=0..15, lanes 16-31: K=16..31)
  v8u Hu = {0,0,0,0,0,0,0,0};

  // Core RNN step: updates H, leaves tanh'd hidden values in t0/t1.
  auto step_core = [&](float xv, float* t0, float* t1) {
    v8f c0, c1;
#pragma unroll
    for (int r = 0; r < 8; ++r) {
      c0[r] = fmaf(xv, wih0[r], bb0[r]);
      c1[r] = fmaf(xv, wih1[r], bb1[r]);
    }
    v16h Hb = __builtin_bit_cast(v16h, Hu);
    v8f d0 = __builtin_amdgcn_wmma_f32_16x16x32_f16(false, A0, false, Hb, (short)0, c0, false, false);
    v8f d1 = __builtin_amdgcn_wmma_f32_16x16x32_f16(false, A1, false, Hb, (short)0, c1, false, false);
#pragma unroll
    for (int r = 0; r < 8; ++r) { t0[r] = fast_tanh(d0[r]); t1[r] = fast_tanh(d1[r]); }
    // repack D -> B operand: pure per-lane f16 packing, no cross-lane traffic
#pragma unroll
    for (int q = 0; q < 4; ++q) {
      Hu[q]     = pkh2(t0[2 * q], t0[2 * q + 1]);   // K = 0..7   (lo) / 16..23 (hi)
      Hu[4 + q] = pkh2(t1[2 * q], t1[2 * q + 1]);   // K = 8..15  (lo) / 24..31 (hi)
    }
  };

  // Half-dot for the decoder, 4-way split accumulators (dep depth 4+2, not 16)
  auto decode_part = [&](const float* t0, const float* t1) -> float {
    float a0 = 0.f, a1 = 0.f, a2 = 0.f, a3 = 0.f;
#pragma unroll
    for (int r = 0; r < 8; ++r) {
      float* a = (r & 2) ? ((r & 1) ? &a3 : &a2) : ((r & 1) ? &a1 : &a0);
      *a = fmaf(t0[r], wd0[r], *a);
    }
#pragma unroll
    for (int r = 0; r < 8; ++r) {
      float* a = (r & 2) ? ((r & 1) ? &a3 : &a2) : ((r & 1) ? &a1 : &a0);
      *a = fmaf(t1[r], wd1[r], *a);
    }
    return (a0 + a1) + (a2 + a3);
  };

  // ================= main scan over T, in LDS-staged chunks =================
  for (int tq = 0; tq < T_MAIN; tq += TC) {
#pragma unroll 1
    for (int i = lane; i < 16 * TC; i += 32) {          // coalesced x tile load
      int nn = i / TC, tt = i % TC;
      xs[nn * XSTR + tt] = x[(size_t)(b0 + nn) * T_MAIN + tq + tt];
    }
    __syncthreads();
    // software-pipelined x: consume value loaded last iteration, prefetch next.
    float xv = xs[n * XSTR];
#pragma unroll 2
    for (int tt = 0; tt < TC; ++tt) {
      float xnext = xs[n * XSTR + tt + 1];              // pad read at tt=TC-1 (dead)
      float t0[8], t1[8];
      step_core(xv, t0, t1);
      // per-lane half-dot; halves combined at flush -> no swizzle, no branch
      ys2[lane * YSTR2 + tt] = decode_part(t0, t1);
      xv = xnext;
    }
    __syncthreads();
#pragma unroll 1
    for (int i = lane; i < 16 * TC; i += 32) {          // coalesced y tile store
      int nn = i / TC, tt = i % TC;
      float v = ys2[nn * YSTR2 + tt] + ys2[(16 + nn) * YSTR2 + tt] + bdec;
      y[(size_t)(b0 + nn) * T_OUT + tq + tt] = v;
    }
    __syncthreads();
  }

  // capture the last-8 window for the autoregressive tail (ys2 = last chunk)
#pragma unroll 1
  for (int i = lane; i < 16 * T_EXTW; i += 32) {
    int nn = i / T_EXTW, k = i % T_EXTW;
    int tt = (TC - T_EXTW) + k;
    ybuf[nn * PSTR + k] = ys2[nn * YSTR2 + tt] + ys2[(16 + nn) * YSTR2 + tt] + bdec;
  }
  __syncthreads();

  // ================= autoregressive tail: 16 predictions =================
#pragma unroll 1
  for (int s = 0; s < T_PRED; ++s) {
    Hu = (v8u){0,0,0,0,0,0,0,0};                        // h0 = 0 each rollout
    float t0[8], t1[8];
    // fully unrolled: all 8 window loads can issue before the serial chain
#pragma unroll
    for (int k = 0; k < T_EXTW; ++k) {
      float xv = ybuf[n * PSTR + s + k];                // sliding 8-wide window
      step_core(xv, t0, t1);
    }
    // only the final step of the rollout is decoded
    float part = decode_part(t0, t1);
    part += swap16f(part);                              // both halves -> same value
    ybuf[n * PSTR + T_EXTW + s] = part + bdec;          // duplicate write, same value
    __syncthreads();
  }
#pragma unroll 1
  for (int i = lane; i < 16 * T_PRED; i += 32) {
    int nn = i / T_PRED, cc = i % T_PRED;
    y[(size_t)(b0 + nn) * T_OUT + T_MAIN + cc] = ybuf[nn * PSTR + T_EXTW + cc];
  }
}

extern "C" void kernel_launch(void* const* d_in, const int* in_sizes, int n_in,
                              void* d_out, int out_size, void* d_ws, size_t ws_size,
                              hipStream_t stream) {
  const float* x     = (const float*)d_in[0];
  const float* W_ih  = (const float*)d_in[1];
  const float* b_ih  = (const float*)d_in[2];
  const float* W_hh  = (const float*)d_in[3];
  const float* b_hh  = (const float*)d_in[4];
  const float* W_dec = (const float*)d_in[5];
  const float* b_dec = (const float*)d_in[6];
  float* y = (float*)d_out;

  dim3 grid(B_TOT / 16), block(32);
  hipLaunchKernelGGL(rnn_wave_kernel, grid, block, 0, stream,
                     x, W_ih, b_ih, W_hh, b_hh, W_dec, b_dec, y);
}
